// Head_14431090115025
// MI455X (gfx1250) — matrix operand order repaired and checked by reading, not compiled, not run on hardware
//
#include <hip/hip_runtime.h>
#include <hip/hip_bf16.h>
#include <stddef.h>

// ---------------------------------------------------------------------------
// Single-head causal attention for MI455X (gfx1250, wave32, WMMA bf16).
//   B=256, T=256, C=768, H=64
//   kernel 1: pack Wq/Wk/Wv into WMMA B-fragment order (bf16)
//   kernel 2: QKV projection GEMM via v_wmma_f32_16x16x32_bf16 (V stored
//             transposed per batch so attention's PV B-fragments are
//             contiguous in LDS)
//   kernel 3: per-batch flash attention, K/V staged to LDS via the Tensor
//             Data Mover (tensor_load_to_lds + s_wait_tensorcnt).
// f32->bf16 uses truncation: packing via a single v_perm_b32, scalar stores
// via d16_hi stores (zero VALU cost).
// ---------------------------------------------------------------------------

typedef __attribute__((ext_vector_type(16))) __bf16 v16bf;
typedef __attribute__((ext_vector_type(8)))  float  v8f;
typedef __attribute__((ext_vector_type(4)))  unsigned v4u;
typedef __attribute__((ext_vector_type(8)))  int    v8i;
typedef __attribute__((ext_vector_type(4)))  int    v4i;

#define N_EMB 768
#define TT 256
#define BB 256
#define HS 64

union BF16x16 {
    unsigned short u[16];
    unsigned int   u32[8];
    v16bf          v;
};

// pack {bf16(y), bf16(x)} (truncation) with one v_perm_b32
static __device__ __forceinline__ unsigned pk_bf16(float x, float y) {
    return __builtin_amdgcn_perm(__float_as_uint(y), __float_as_uint(x),
                                 0x07060302u);
}

// truncating bf16 store: backend emits a d16_hi 16-bit store (no VALU)
static __device__ __forceinline__ void store_bf16(__bf16* p, float f) {
    *(unsigned short*)p = (unsigned short)(__float_as_uint(f) >> 16);
}

// ---------------------------------------------------------------------------
// Tensor Data Mover: 1-D descriptor load of `nelem` 2-byte elements into LDS.
// D# per CDNA5 ISA 8.3/8.4: group0 {count=1, lds_addr, global_addr, type=2},
// group1 {data_size=1(2B), tensor_dim0=nelem, tensor_dim1=1, tile_dim0=nelem,
// stride=nelem}.
// ---------------------------------------------------------------------------
#if __has_builtin(__builtin_amdgcn_tensor_load_to_lds)
#define HAVE_TDM 1
static __device__ __forceinline__ void tdm_load_lds_1d(unsigned ldsOff,
                                                       const void* src,
                                                       unsigned nelem) {
    unsigned long long ga = (unsigned long long)(size_t)src;
    v4u g0 = {1u,                                       // count=1, user mode
              ldsOff,                                   // lds_addr (bytes)
              (unsigned)ga,                             // global_addr[31:0]
              ((unsigned)(ga >> 32) & 0x01FFFFFFu) | 0x80000000u}; // type=2
    v8i g1 = {(int)(1u << 16),                          // data_size=1 (2B)
              (int)((nelem & 0xFFFFu) << 16),           // tensor_dim0 lo16
              (int)(((nelem >> 16) & 0xFFFFu) | (1u << 16)), // dim0 hi, dim1=1
              (int)((nelem & 0xFFFFu) << 16),           // tile_dim0 = nelem
              0,                                        // tile_dim1/2 = 0
              (int)nelem,                               // dim0_stride lo32
              0, 0};
    v4i z4 = {0, 0, 0, 0};
#if defined(__clang_major__) && __clang_major__ >= 23
    v8i z8 = {0, 0, 0, 0, 0, 0, 0, 0};
    __builtin_amdgcn_tensor_load_to_lds(g0, g1, z4, z4, z8, 0);
#else
    __builtin_amdgcn_tensor_load_to_lds(g0, g1, z4, z4, 0);
#endif
}
#endif

// B-fragment K index (16x16x32 bf16): K_B(e) = ((e&8)<<1) + half*8 + (e&7)
// A-fragment K index:                 K_A(e) = ((e&8)<<1) + half*8 + (e&7)

// ---------------------------------------------------------------------------
// Kernel 1: pack weights into B-fragment-ordered bf16.
// frag f = (mtx*24 + kbIdx)*4 + nt ; storage pw[f*512 + lane*16 + e]
// ---------------------------------------------------------------------------
__global__ void packW_kernel(const float* __restrict__ Wq,
                             const float* __restrict__ Wk,
                             const float* __restrict__ Wv,
                             __bf16* __restrict__ pw) {
    int tid = blockIdx.x * blockDim.x + threadIdx.x;
    if (tid >= 288 * 32) return;
    int lane = tid & 31;
    int frag = tid >> 5;
    int nt = frag & 3;
    int t2 = frag >> 2;
    int kbIdx = t2 % 24;
    int mtx = t2 / 24;
    const float* W = (mtx == 0) ? Wq : ((mtx == 1) ? Wk : Wv);
    int half = lane >> 4, l16 = lane & 15;
    int n = nt * 16 + l16;
    int kb = kbIdx * 32;
    __bf16* dst = pw + (size_t)frag * 512 + (size_t)lane * 16;
#pragma unroll
    for (int e = 0; e < 16; ++e) {
        int k = kb + ((e & 8) << 1) + half * 8 + (e & 7);
        store_bf16(dst + e, W[(size_t)k * HS + n]);
    }
}

// ---------------------------------------------------------------------------
// Kernel 2: QKV projection. One wave = one 16-row tile of x, 12 accumulators.
// Q,K stored [token][head]; V stored transposed per batch: [head][token].
// ---------------------------------------------------------------------------
__global__ void qkv_kernel(const float* __restrict__ x,
                           const __bf16* __restrict__ pw,
                           __bf16* __restrict__ Qo,
                           __bf16* __restrict__ Ko,
                           __bf16* __restrict__ Vt) {
    const int lane = threadIdx.x & 31;
    const int wave = threadIdx.x >> 5;
    const int tile = blockIdx.x * 8 + wave;
    const int m0 = tile * 16;
    const int half = lane >> 4;
    const int l16 = lane & 15;

    v8f acc[12];
#pragma unroll
    for (int t = 0; t < 12; ++t) acc[t] = (v8f){};

    for (int kb = 0; kb < N_EMB; kb += 32) {
        const float* xrow = x + (size_t)(m0 + l16) * N_EMB + kb + half * 8;
        __builtin_prefetch(xrow + 32, 0, 3);
        // A fragment: K_A runs of 8 are contiguous -> 4 x float4 loads,
        // one v_perm_b32 per bf16 pair
        BF16x16 a;
        {
            float4 f0 = *(const float4*)(xrow);       // e 0..3
            float4 f1 = *(const float4*)(xrow + 4);   // e 4..7
            float4 f2 = *(const float4*)(xrow + 16);  // e 8..11
            float4 f3 = *(const float4*)(xrow + 20);  // e 12..15
            a.u32[0] = pk_bf16(f0.x, f0.y);
            a.u32[1] = pk_bf16(f0.z, f0.w);
            a.u32[2] = pk_bf16(f1.x, f1.y);
            a.u32[3] = pk_bf16(f1.z, f1.w);
            a.u32[4] = pk_bf16(f2.x, f2.y);
            a.u32[5] = pk_bf16(f2.z, f2.w);
            a.u32[6] = pk_bf16(f3.x, f3.y);
            a.u32[7] = pk_bf16(f3.z, f3.w);
        }
        const int kbIdx = kb >> 5;
#pragma unroll
        for (int mtx = 0; mtx < 3; ++mtx) {
#pragma unroll
            for (int nt = 0; nt < 4; ++nt) {
                const int f = (mtx * 24 + kbIdx) * 4 + nt;
                const v16bf bv =
                    *(const v16bf*)(pw + (size_t)f * 512 + (size_t)lane * 16);
                acc[mtx * 4 + nt] = __builtin_amdgcn_wmma_f32_16x16x32_bf16(
                    false, a.v, false, bv, (short)0, acc[mtx * 4 + nt], false, false);
            }
        }
    }

    // Q, K: [token][head]
#pragma unroll
    for (int mtx = 0; mtx < 2; ++mtx) {
        __bf16* O = (mtx == 0) ? Qo : Ko;
#pragma unroll
        for (int nt = 0; nt < 4; ++nt) {
            v8f c = acc[mtx * 4 + nt];
#pragma unroll
            for (int i = 0; i < 8; ++i) {
                int row = m0 + half * 8 + i;
                int col = nt * 16 + l16;
                store_bf16(O + (size_t)row * HS + col, c[i]);
            }
        }
    }
    // V: transposed per batch: Vt[b*16384 + head*256 + token]
#pragma unroll
    for (int nt = 0; nt < 4; ++nt) {
        v8f c = acc[8 + nt];
#pragma unroll
        for (int i = 0; i < 8; ++i) {
            int row = m0 + half * 8 + i;        // global token index
            int b = row >> 8;
            int tok = row & 255;
            int col = nt * 16 + l16;            // head dim
            store_bf16(Vt + (size_t)b * (TT * HS) + (size_t)col * TT + tok, c[i]);
        }
    }
}

// ---------------------------------------------------------------------------
// Kernel 3: attention. One block (8 waves) per batch. K ([tok][head]) and
// Vt ([head][tok]) resident in LDS; TDM DMA stage when available.
// ---------------------------------------------------------------------------
__global__ void attn_kernel(const __bf16* __restrict__ Qw,
                            const __bf16* __restrict__ Kw,
                            const __bf16* __restrict__ Vtw,
                            float* __restrict__ out) {
    extern __shared__ __align__(16) char smem[];
    __bf16* kbuf   = (__bf16*)smem;                 // 256*64  [tok][head]
    __bf16* vbuf   = kbuf + TT * HS;                // 64*256  [head][tok]
    __bf16* pstage = vbuf + TT * HS;                // 8 waves * 16*32

    const int b    = blockIdx.x;
    const int lane = threadIdx.x & 31;
    const int wave = threadIdx.x >> 5;
    const int half = lane >> 4;
    const int l16  = lane & 15;
    const float scale = 0.125f;  // 1/sqrt(64)

#if defined(HAVE_TDM)
    if (wave == 0) {
        tdm_load_lds_1d(0u, Kw + (size_t)b * TT * HS, TT * HS);
        tdm_load_lds_1d((unsigned)(TT * HS * sizeof(unsigned short)),
                        Vtw + (size_t)b * TT * HS, TT * HS);
        __builtin_amdgcn_s_wait_tensorcnt(0);
    }
    __syncthreads();
#else
    {
        const uint4* srcK = (const uint4*)(Kw + (size_t)b * TT * HS);
        const uint4* srcV = (const uint4*)(Vtw + (size_t)b * TT * HS);
        uint4* dK = (uint4*)kbuf;
        uint4* dV = (uint4*)vbuf;
        for (int i = threadIdx.x; i < (TT * HS) / 8; i += 256) {
            dK[i] = srcK[i];
            dV[i] = srcV[i];
        }
    }
    __syncthreads();
#endif

#pragma unroll
    for (int pass = 0; pass < 2; ++pass) {
        const int qt = (pass == 0) ? wave : (15 - wave);   // causal load balance
        const int qRow = qt * 16;

        // Q A-fragments (head dim 0..31 / 32..63)
        BF16x16 qf[2];
        const __bf16* qrow = Qw + ((size_t)b * TT + qRow + l16) * HS;
#pragma unroll
        for (int f = 0; f < 2; ++f) {
#pragma unroll
            for (int j = 0; j < 8; ++j) {
                int e = 2 * j;
                int ko = half * 16 + e;
                qf[f].u32[j] = *(const unsigned*)(qrow + f * 32 + ko);
            }
        }

        v8f o0 = (v8f){}, o1 = (v8f){}, o2 = (v8f){}, o3 = (v8f){};
        float m[8], l[8];
#pragma unroll
        for (int i = 0; i < 8; ++i) { m[i] = -1e30f; l[i] = 0.f; }

        for (int kb = 0; kb < qRow + 16; kb += 32) {
            // ---- scores: S[16 x 32] = Q(16x64) * K^T, 4 WMMAs ----
            v8f s0 = (v8f){}, s1 = (v8f){};
#pragma unroll
            for (int kf = 0; kf < 2; ++kf) {
                const v16bf b0 = *(const v16bf*)(kbuf +
                    (size_t)(kb + l16) * HS + kf * 32 + half * 16);
                s0 = __builtin_amdgcn_wmma_f32_16x16x32_bf16(
                    false, qf[kf].v, false, b0, (short)0, s0, false, false);
                const v16bf b1 = *(const v16bf*)(kbuf +
                    (size_t)(kb + 16 + l16) * HS + kf * 32 + half * 16);
                s1 = __builtin_amdgcn_wmma_f32_16x16x32_bf16(
                    false, qf[kf].v, false, b1, (short)0, s1, false, false);
            }

            // ---- scale + causal mask + online softmax update ----
            const int key0 = kb + l16;
            const int key1 = kb + 16 + l16;
            const int pbase = wave * 512;
#pragma unroll
            for (int i = 0; i < 8; ++i) {
                const int r = qRow + half * 8 + i;
                float a0 = s0[i] * scale;
                float a1 = s1[i] * scale;
                if (key0 > r) a0 = -1e30f;
                if (key1 > r) a1 = -1e30f;
                float cm = fmaxf(a0, a1);
                cm = fmaxf(cm, __shfl_xor(cm, 1));
                cm = fmaxf(cm, __shfl_xor(cm, 2));
                cm = fmaxf(cm, __shfl_xor(cm, 4));
                cm = fmaxf(cm, __shfl_xor(cm, 8));
                const float mn = fmaxf(m[i], cm);
                const float alpha = __expf(m[i] - mn);
                const float p0 = __expf(a0 - mn);
                const float p1 = __expf(a1 - mn);
                float rs = p0 + p1;
                rs += __shfl_xor(rs, 1);
                rs += __shfl_xor(rs, 2);
                rs += __shfl_xor(rs, 4);
                rs += __shfl_xor(rs, 8);
                l[i] = l[i] * alpha + rs;
                m[i] = mn;
                o0[i] *= alpha; o1[i] *= alpha; o2[i] *= alpha; o3[i] *= alpha;
                const int rl = half * 8 + i;
                store_bf16(&pstage[pbase + rl * 32 + l16], p0);
                store_bf16(&pstage[pbase + rl * 32 + 16 + l16], p1);
            }
            // wave-private LDS round trip: make the store data visible
            asm volatile("s_wait_dscnt 0" ::: "memory");

            // ---- P as A-fragment (C-layout -> A-layout via LDS) ----
            BF16x16 pf;
#pragma unroll
            for (int j = 0; j < 8; ++j) {
                int e = 2 * j;
                int ko = half * 16 + e;
                pf.u32[j] = *(const unsigned*)(pstage + pbase + l16 * 32 + ko);
            }

            // ---- O += P(16x32) * V(32x64): 4 WMMAs, contiguous Vt frags ----
#pragma unroll
            for (int t = 0; t < 4; ++t) {
                const v16bf vfv = *(const v16bf*)(vbuf +
                    (size_t)(t * 16 + l16) * TT + kb + half * 16);
                v8f* op = (t == 0) ? &o0 : (t == 1) ? &o1 : (t == 2) ? &o2 : &o3;
                *op = __builtin_amdgcn_wmma_f32_16x16x32_bf16(
                    false, pf.v, false, vfv, (short)0, *op, false, false);
            }
        }

        // ---- normalize + store fp32 output ----
#pragma unroll
        for (int i = 0; i < 8; ++i) {
            const float inv = 1.0f / l[i];
            const size_t row = (size_t)b * TT + qRow + half * 8 + i;
            out[row * HS + 0 * 16 + l16] = o0[i] * inv;
            out[row * HS + 1 * 16 + l16] = o1[i] * inv;
            out[row * HS + 2 * 16 + l16] = o2[i] * inv;
            out[row * HS + 3 * 16 + l16] = o3[i] * inv;
        }
    }
}

// ---------------------------------------------------------------------------
extern "C" void kernel_launch(void* const* d_in, const int* in_sizes, int n_in,
                              void* d_out, int out_size, void* d_ws, size_t ws_size,
                              hipStream_t stream) {
    const float* x  = (const float*)d_in[0];
    const float* Wq = (const float*)d_in[1];
    const float* Wk = (const float*)d_in[2];
    const float* Wv = (const float*)d_in[3];

    char* ws = (char*)d_ws;
    const size_t qkvBytes = (size_t)BB * TT * HS * sizeof(unsigned short); // 8 MB each
    __bf16* Q  = (__bf16*)(ws);
    __bf16* K  = (__bf16*)(ws + qkvBytes);
    __bf16* Vt = (__bf16*)(ws + 2 * qkvBytes);
    __bf16* PW = (__bf16*)(ws + 3 * qkvBytes);   // 288 frags * 512 bf16 = 288 KB

    packW_kernel<<<36, 256, 0, stream>>>(Wq, Wk, Wv, PW);
    qkv_kernel<<<512, 256, 0, stream>>>(x, PW, Q, K, Vt);

    const size_t smemBytes = (2 * (size_t)TT * HS + 8 * 512) * sizeof(unsigned short); // 72 KB
    attn_kernel<<<BB, 256, smemBytes, stream>>>(Q, K, Vt, (float*)d_out);
}
